// ResidualEmbedding_70798240907390
// MI455X (gfx1250) — compile-verified, matrix operand order checked by reading, not run on hardware
//
#include <hip/hip_runtime.h>

#define FRAME_   160
#define KFR      15
#define ORDER_   12
#define WAVES_PB 8
#define TPB      (WAVES_PB * 32)

#if defined(__has_builtin)
#if __has_builtin(__builtin_amdgcn_global_load_async_to_lds_b32)
#define HAVE_ASYNC_LD 1
#endif
#if __has_builtin(__builtin_amdgcn_global_store_async_from_lds_b32)
#define HAVE_ASYNC_ST 1
#endif
#endif

typedef __attribute__((ext_vector_type(16))) _Float16 v16h;
typedef __attribute__((ext_vector_type(8)))  float    v8f;

typedef __attribute__((address_space(1))) int gas_int;
typedef __attribute__((address_space(3))) int las_int;

__device__ __forceinline__ float wave_sum(float v) {
#pragma unroll
  for (int s = 16; s > 0; s >>= 1) v += __shfl_xor(v, s, 32);
  return v;
}

__global__ __launch_bounds__(TPB) void residual_embed_70798240907390(
    const int* __restrict__ bits, const float* __restrict__ pcm,
    const float* __restrict__ alphap, float* __restrict__ out) {
  // Per-wave private LDS slices: raw frame X, Burg forward F, backward Bv, coeffs A.
  // sB doubles as the output staging buffer after the recursion.
  __shared__ float sX[WAVES_PB][FRAME_ + 8];
  __shared__ float sF[WAVES_PB][FRAME_ + 8];
  __shared__ float sB[WAVES_PB][FRAME_ + 8];
  __shared__ float sA[WAVES_PB][16];

  const int lane  = threadIdx.x & 31;
  const int wv    = threadIdx.x >> 5;
  const int frame = blockIdx.x * WAVES_PB + wv;   // grid sized exactly: no tail
  const int bidx  = frame / KFR;
  const int kidx  = frame - bidx * KFR;

  float* X  = sX[wv];
  float* F  = sF[wv];
  float* Bv = sB[wv];
  float* A  = sA[wv];

  const size_t base = (size_t)frame * FRAME_;     // == bidx*T + kidx*FRAME
  const float* src  = pcm + base;

  __builtin_prefetch(src, 0, 3);   // global_prefetch_b8

  // ---- stage raw frame into LDS (CDNA5 async global->LDS) ----
#ifdef HAVE_ASYNC_LD
#pragma unroll
  for (int m = 0; m < 5; ++m) {
    int n = lane + 32 * m;
    __builtin_amdgcn_global_load_async_to_lds_b32(
        (gas_int*)(src + n), (las_int*)(X + n), 0, 0);
  }
  asm volatile("s_wait_asynccnt 0" ::: "memory");
#else
#pragma unroll
  for (int m = 0; m < 5; ++m) {
    int n = lane + 32 * m;
    X[n] = src[n];
  }
#endif

  // ---- window (np.hanning) and init Burg arrays; keep raw samples in regs ----
  float xr[5];
#pragma unroll
  for (int m = 0; m < 5; ++m) {
    int n = lane + 32 * m;
    float xv = X[n];
    xr[m] = xv;
    float wn = 0.5f - 0.5f * cosf((float)n * (6.283185307179586f / 159.0f));
    float wx = xv * wn;
    F[n]  = wx;   // f = wx[1:]  -> f[j] = F[i+1+j]
    Bv[n] = wx;   // b = wx[:-1] -> b[j] = Bv[j]
    // den = sum(f*f + b*b): element n counted twice except endpoints
    xv = wx * wx;
    xr[m] = xr[m];   // keep raw
    (void)xv;
  }
  if (lane < 16) A[lane] = (lane == 0) ? 1.0f : 0.0f;

  float acc = 0.0f;
#pragma unroll
  for (int m = 0; m < 5; ++m) {
    int n = lane + 32 * m;
    float wn = 0.5f - 0.5f * cosf((float)n * (6.283185307179586f / 159.0f));
    float wx = xr[m] * wn;
    float c = wx * wx;
    acc += ((n >= 1) ? c : 0.0f) + ((n <= 158) ? c : 0.0f);
  }
  float den = wave_sum(acc);

  // ---- Burg recursion: 12 fully-unrolled iterations, one wave per frame ----
#pragma unroll
  for (int i = 0; i < ORDER_; ++i) {
    const int L = 159 - i;          // compile-time
    float fj[5], bj[5];
    float dot = 0.0f;
#pragma unroll
    for (int m = 0; m < 4; ++m) {   // j = lane + 32m <= 127 < L always
      int j = lane + 32 * m;
      fj[m] = F[i + 1 + j];
      bj[m] = Bv[j];
      dot += fj[m] * bj[m];
    }
    {                               // tail: j = lane + 128, valid iff lane < 31 - i
      int j = lane + 128;
      if (lane < 31 - i) {
        fj[4] = F[i + 1 + j];
        bj[4] = Bv[j];
        dot += fj[4] * bj[4];
      } else {
        fj[4] = 0.0f;
        bj[4] = 0.0f;
      }
    }
    float num = wave_sum(dot);
    float r = -2.0f * num / den;    // wave-uniform

    // a[:i+2] += r * a[:i+2][::-1]  (reads precede write; in-order LDS per wave)
    if (lane <= i + 1) {
      float a0 = A[lane];
      float a1 = A[i + 1 - lane];
      A[lane] = a0 + r * a1;
    }

    // f += r*b ; b += r*f_old
#pragma unroll
    for (int m = 0; m < 4; ++m) {
      int j = lane + 32 * m;
      F[i + 1 + j] = fj[m] + r * bj[m];
      Bv[j]        = bj[m] + r * fj[m];
    }
    if (lane < 31 - i) {
      int j = lane + 128;
      F[i + 1 + j] = fj[4] + r * bj[4];
      Bv[j]        = bj[4] + r * fj[4];
    }

    // den = (1-r^2)*den - b_new[L-1]^2 - f_new[0]^2
    // f_new[0]  lives in lane 0 (j=0, m=0); b_new[L-1] in lane 30-i (j=L-1, m=4).
    float cf0 = fj[0] + r * bj[0];
    float cbl = bj[4] + r * fj[4];
    float f0 = __shfl(cf0, 0, 32);
    float bl = __shfl(cbl, 30 - i, 32);
    den = (1.0f - r * r) * den - bl * bl - f0 * f0;
  }

  // ---- FIR as one WMMA: D(p,c) = sum_k a[k] * x[16p + c - k] ----
  // A(p,j) = x[16p + j - 16]  (16x32 f16, rows p>=10 zeroed)
  // B(j,c) = a[c - j + 16]    (Toeplitz of LPC coeffs, zero outside 0..12)
  const int cN  = lane & 15;   // A row (M) and B column (N)
  const int grp = lane >> 4;   // K-half select per ISA 16-bit layout
  v16h av, bvv;
#pragma unroll
  for (int h = 0; h < 16; ++h) {
    int v = h >> 1, t = h & 1;
    int Kk = (v < 4) ? (2 * v + t) : (16 + 2 * (v - 4) + t);
    Kk += grp << 3;
    float va = 0.0f;
    if (cN < 10) {
      int xi = (cN << 4) + Kk - 16;
      if (xi >= 0 && xi < FRAME_) va = X[xi];
    }
    av[h] = (_Float16)va;
    float vb = 0.0f;
    int ai = cN - Kk + 16;
    if (ai >= 0 && ai <= ORDER_) vb = A[ai];
    bvv[h] = (_Float16)vb;
  }
  v8f dmat = {};
  dmat = __builtin_amdgcn_wmma_f32_16x16x32_f16(false, av, false, bvv,
                                                (short)0, dmat, false, false);

  // ---- combine: out = pcm + alpha * res * (2*bit - 1) ----
  const float alpha = alphap[0];
  const float sgn = (float)(2 * bits[bidx * KFR + kidx] - 1);
  const float asn = alpha * sgn;
  float* dst = out + base;
  float* W   = Bv;             // reuse as output staging buffer

#ifdef HAVE_ASYNC_ST
  // Scatter final samples into LDS (D-layout -> linear), then async LDS->global.
#pragma unroll
  for (int d = 0; d < 8; ++d) {
    int M = d + (grp << 3);    // D: lanes 0-15 -> M=d, lanes 16-31 -> M=d+8
    if (M < 10) {
      int n = (M << 4) + cN;
      W[n] = X[n] + asn * dmat[d];
    }
  }
  asm volatile("s_wait_dscnt 0" ::: "memory");   // LDS visible to async engine
#pragma unroll
  for (int m = 0; m < 5; ++m) {
    int n = lane + 32 * m;
    __builtin_amdgcn_global_store_async_from_lds_b32(
        (gas_int*)(dst + n), (las_int*)(W + n), 0, 0);
  }
  asm volatile("s_wait_asynccnt 0" ::: "memory");
#else
#pragma unroll
  for (int d = 0; d < 8; ++d) {
    int M = d + (grp << 3);
    if (M < 10) {
      int n = (M << 4) + cN;
      __builtin_nontemporal_store(X[n] + asn * dmat[d], dst + n);
    }
  }
#endif
}

extern "C" void kernel_launch(void* const* d_in, const int* in_sizes, int n_in,
                              void* d_out, int out_size, void* d_ws, size_t ws_size,
                              hipStream_t stream) {
  (void)n_in; (void)out_size; (void)d_ws; (void)ws_size;
  const int*   bits  = (const int*)d_in[0];
  const float* pcm   = (const float*)d_in[1];
  const float* alpha = (const float*)d_in[2];
  float*       out   = (float*)d_out;

  const int frames = in_sizes[0];          // B*K = 61440, divisible by WAVES_PB
  const int blocks = frames / WAVES_PB;
  residual_embed_70798240907390<<<dim3(blocks), dim3(TPB), 0, stream>>>(
      bits, pcm, alpha, out);
}
